// PianoTreeDecoder_21002390078095
// MI455X (gfx1250) — matrix operand order repaired and checked by reading, not compiled, not run on hardware
//
#include <hip/hip_runtime.h>
#include <hip/hip_bf16.h>

typedef __bf16 bf16;
typedef __attribute__((ext_vector_type(16))) __bf16 bf16x16;
typedef __attribute__((ext_vector_type(8)))  __bf16 bf16x8;
typedef __attribute__((ext_vector_type(8)))  float  floatx8;
typedef __attribute__((ext_vector_type(4)))  unsigned int u32x4;
typedef __attribute__((ext_vector_type(8)))  int i32x8;
typedef __attribute__((ext_vector_type(4)))  int i32x4;

#define DEV __device__ __forceinline__

// ---------------- WMMA helpers (CDNA5: v_wmma_f32_16x16x32_bf16) ----------
DEV floatx8 wmma_bf(bf16x16 a, bf16x16 b, floatx8 c) {
  return __builtin_amdgcn_wmma_f32_16x16x32_bf16(false, a, false, b, (short)0, c,
                                                 false, false);
}

// B fragment: pre-swizzled weights, 512 bf16 per 16(N)x32(K) tile, lane*16 contiguous.
DEV bf16x16 load_b(const bf16* pk, int tile, int lane) {
  return *(const bf16x16*)(pk + ((size_t)tile << 9) + (lane << 4));
}

// A fragment from LDS bf16 row-major [16][ldk]; 16-bit A layout:
//   m = lane&15 ; k = j%8 + 16*(j/8) + 8*(lane/16)  => two contiguous 16B runs.
DEV bf16x16 load_a_lds(const bf16* base, int ldk, int lane) {
  const int m = lane & 15, k0 = (lane >> 4) << 3;
  const bf16* r = base + m * ldk + k0;
  bf16x8 lo = *(const bf16x8*)(r);
  bf16x8 hi = *(const bf16x8*)(r + 16);
  return __builtin_shufflevector(lo, hi, 0, 1, 2, 3, 4, 5, 6, 7,
                                 8, 9, 10, 11, 12, 13, 14, 15);
}

// A fragment gathered from global f32 row-major (z at startup).
DEV bf16x16 load_a_g32(const float* base, int ldk, int lane) {
  const int m = lane & 15, k0 = (lane >> 4) << 3;
  const float* r = base + (size_t)m * ldk + k0;
  bf16x16 a;
#pragma unroll
  for (int j = 0; j < 8; ++j) { a[j] = (bf16)r[j]; a[j + 8] = (bf16)r[16 + j]; }
  return a;
}

// C/D layout: VGPR r, lane L -> M = r + 8*(L/16), N = L%16
DEV void store_c(float* base, int ldn, int lane, floatx8 c) {
  const int n = lane & 15, m0 = (lane >> 4) << 3;
#pragma unroll
  for (int r = 0; r < 8; ++r) base[(m0 + r) * ldn + n] = c[r];
}

DEV void store_c_bf_bias(bf16* base, int ldn, int lane, floatx8 c, const float* bias) {
  const int n = lane & 15, m0 = (lane >> 4) << 3;
  const float bv = bias[n];
#pragma unroll
  for (int r = 0; r < 8; ++r) base[(m0 + r) * ldn + n] = (bf16)(c[r] + bv);
}

DEV float sigm(float x) { return 1.0f / (1.0f + __expf(-x)); }

// ---------------- Tensor Data Mover: 1-D global->LDS DMA (ISA ch.8 D#) -----
DEV void tdm_load_1d(const float* gsrc, unsigned lds_byte_off, int n) {
  unsigned long long ga = (unsigned long long)(uintptr_t)gsrc;
  u32x4 g0 = (u32x4)0u;
  g0[0] = 1u;                                       // count=1 (valid), user mode
  g0[1] = lds_byte_off;                             // lds_addr
  g0[2] = (unsigned)(ga & 0xFFFFFFFFu);             // global_addr[31:0]
  g0[3] = (unsigned)((ga >> 32) & 0x1FFFFFFu) | (2u << 30);  // addr[56:32]|type=2
  i32x8 g1 = (i32x8)0;
  g1[0] = 0x20000;                                  // data_size=2 (4 bytes)
  g1[1] = (n & 0xFFFF) << 16;                       // tensor_dim0[15:0]
  g1[2] = ((n >> 16) & 0xFFFF) | (1 << 16);         // tensor_dim0[31:16]|dim1=1
  g1[3] = (n & 0xFFFF) << 16;                       // tile_dim0 = n
  g1[4] = 1;                                        // tile_dim1 = 1
  g1[5] = n;                                        // tensor_dim0_stride
  i32x4 g2 = (i32x4)0, g3 = (i32x4)0;
#if __clang_major__ >= 23
  __builtin_amdgcn_tensor_load_to_lds(g0, g1, g2, g3, (i32x8)0, 0);
#else
  __builtin_amdgcn_tensor_load_to_lds(g0, g1, g2, g3, 0);
#endif
}

// ---------------- problem constants ----------------
#define NB       512
#define ZK       512
#define TH       1024     // time hidden
#define NH       512      // notes hidden
#define EH       128      // emb hidden
#define NE       128      // note emb
#define ZI       256      // z_in
#define NSTEP    32
#define NINNER   15
#define PR       130      // pitch range
#define PRP      144      // padded
#define NTK      160      // padded note token K (135 -> 160)

// ---------------- LDS layout ----------------
constexpr int SM_HT   = 0;                        // bf16 [2][16][1024]
constexpr int SM_HN   = SM_HT + 2 * 16 * TH;      // bf16 [2][16][512]
constexpr int SM_ZIN  = SM_HN + 2 * 16 * NH;      // bf16 [16][256]
constexpr int SM_TOK  = SM_ZIN + 16 * ZI;         // bf16 [16][256]
constexpr int SM_PRED = SM_TOK + 16 * ZI;         // bf16 [16(t)][16][128]
constexpr int SM_NTOK = SM_PRED + 16 * 16 * NE;   // bf16 [16][160]
constexpr int SM_HE   = SM_NTOK + 16 * NTK;       // bf16 [2dir][2buf][16][128]
constexpr int SM_SOS  = SM_HE + 2 * 2 * 16 * EH;  // bf16 [128]
constexpr int SM_BF_END = SM_SOS + NE;            // bf16 elems
// float region (offsets in floats)
constexpr int SM_F_GI   = 0;                      // 6144 ([3][16][128] or [16][384])
constexpr int SM_F_GH   = 6144;                   // 6144
constexpr int SM_F_HEAD = 12288;                  // [16][144]
constexpr int SM_F_DUR  = SM_F_HEAD + 16 * PRP;   // [16][16]
constexpr int SM_F_BTIH = SM_F_DUR + 256;         // time_gru b_ih [3072]
constexpr int SM_F_BTHH = SM_F_BTIH + 3072;       // time_gru b_hh [3072]
constexpr int SM_F_BNIH = SM_F_BTHH + 3072;       // notes_gru b_ih [1536]
constexpr int SM_F_BNHH = SM_F_BNIH + 1536;       // notes_gru b_hh [1536]
constexpr int SM_F_BE   = SM_F_BNHH + 1536;       // emb biases 4*[384]
constexpr int SM_F_BT2N = SM_F_BE + 4 * 384;      // [512]
constexpr int SM_F_BPIT = SM_F_BT2N + 512;        // [144] (130 loaded)
constexpr int SM_F_BDUR = SM_F_BPIT + PRP;        // [16] (10 loaded)
constexpr int SM_F_BEMB = SM_F_BDUR + 16;         // [128]
constexpr int SM_F_END  = SM_F_BEMB + 128;
constexpr size_t SM_FBASE = (size_t)SM_BF_END * 2;
constexpr size_t SMEM_BYTES = SM_FBASE + (size_t)SM_F_END * 4 + 64;
#define FBYTE(x) ((unsigned)(SM_FBASE + (size_t)(x) * 4))

struct Args {
  const float *z;
  const float *b_z2h, *b_z2i, *b_t2n, *b_pitch, *b_dur, *b_emb;
  const float *nw, *nb;
  const float *t_bih, *t_bhh, *n_bih, *n_bhh;
  const float *ef_bih, *ef_bhh, *eb_bih, *eb_bhh;
  const float *dec_init;
  const bf16 *P_time_ih, *P_time_hh, *P_z2h, *P_z2i, *P_t2n;
  const bf16 *P_notes_ih, *P_notes_hh, *P_pitch, *P_dur, *P_emb;
  const bf16 *P_ef_ih, *P_ef_hh, *P_eb_ih, *P_eb_hh;
  float *out_pitch, *out_dur;
};

// ---------------- weight pre-swizzle: f32 [O,I] -> bf16 B-fragment tiles ----
__global__ void pack_w(const float* __restrict__ W, bf16* __restrict__ P,
                       int O, int I, int Np, int Kp) {
  int idx = blockIdx.x * 256 + threadIdx.x;
  int total = Np * Kp;
  if (idx >= total) return;
  int tile = idx >> 9, r = idx & 511;
  int lane = r >> 4, j = r & 15;
  int nkt = Kp >> 5;
  int nt = tile / nkt, kt = tile - nt * nkt;
  int n = nt * 16 + (lane & 15);
  int k = kt * 32 + ((lane >> 4) << 3) + (j & 7) + ((j >> 3) << 4);
  float v = (n < O && k < I) ? W[(size_t)n * I + k] : 0.0f;
  P[idx] = (bf16)v;
}

// ---------------- fused persistent decoder: 1 workgroup = 16 batch rows -----
__global__ void __launch_bounds__(256)
ptd_main(Args A) {
  const int tid = threadIdx.x, lane = tid & 31, wv = tid >> 5;
  const int b0 = blockIdx.x * 16;

  extern __shared__ char smem[];
  bf16*  bfs  = (bf16*)smem;
  float* fls  = (float*)(smem + SM_FBASE);
  int*   lens = (int*)(smem + SM_FBASE + (size_t)SM_F_END * 4);

  bf16* hT   = bfs + SM_HT;
  bf16* hN   = bfs + SM_HN;
  bf16* zin  = bfs + SM_ZIN;
  bf16* tok  = bfs + SM_TOK;
  bf16* pred = bfs + SM_PRED;
  bf16* ntok = bfs + SM_NTOK;
  bf16* hE   = bfs + SM_HE;
  bf16* sos  = bfs + SM_SOS;
  float* gi    = fls + SM_F_GI;
  float* gh    = fls + SM_F_GH;
  float* head  = fls + SM_F_HEAD;
  float* durs  = fls + SM_F_DUR;
  float* sBtih = fls + SM_F_BTIH;
  float* sBthh = fls + SM_F_BTHH;
  float* sBnih = fls + SM_F_BNIH;
  float* sBnhh = fls + SM_F_BNHH;
  float* sBe   = fls + SM_F_BE;
  float* sBt2n = fls + SM_F_BT2N;
  float* sBpit = fls + SM_F_BPIT;
  float* sBdur = fls + SM_F_BDUR;
  float* sBemb = fls + SM_F_BEMB;

  // ---- stage all recurrent biases into LDS via the Tensor Data Mover ----
  if (wv == 0) {
    tdm_load_1d(A.t_bih,  FBYTE(SM_F_BTIH), 3072);
    tdm_load_1d(A.t_bhh,  FBYTE(SM_F_BTHH), 3072);
    tdm_load_1d(A.n_bih,  FBYTE(SM_F_BNIH), 1536);
    tdm_load_1d(A.n_bhh,  FBYTE(SM_F_BNHH), 1536);
    tdm_load_1d(A.ef_bih, FBYTE(SM_F_BE + 0 * 384), 384);
    tdm_load_1d(A.ef_bhh, FBYTE(SM_F_BE + 1 * 384), 384);
    tdm_load_1d(A.eb_bih, FBYTE(SM_F_BE + 2 * 384), 384);
    tdm_load_1d(A.eb_bhh, FBYTE(SM_F_BE + 3 * 384), 384);
    tdm_load_1d(A.b_t2n,  FBYTE(SM_F_BT2N), 512);
    tdm_load_1d(A.b_pitch, FBYTE(SM_F_BPIT), 130);
    tdm_load_1d(A.b_dur,  FBYTE(SM_F_BDUR), 10);
    tdm_load_1d(A.b_emb,  FBYTE(SM_F_BEMB), 128);
    __builtin_amdgcn_s_wait_tensorcnt(0);
  }

  // ---- SOS embedding: W[:,128] + 2*sum(W[:,130:135]) + b ----
  if (tid < NE) {
    const float* w = A.nw + tid * 135;
    float s = w[128] + 2.0f * (w[130] + w[131] + w[132] + w[133] + w[134]) + A.nb[tid];
    sos[tid] = (bf16)s;
  }
  __syncthreads();
  for (int i = tid; i < 16 * ZI; i += 256) tok[i] = (bf16)A.dec_init[i & 255];
  for (int i = tid; i < 16 * NE; i += 256) pred[i] = sos[i & 127];   // pred[t=0]=sos

  // ---- z_hid -> hT[0]; z_in -> zin (A gathered from global f32 z) ----
  const float* zrow = A.z + (size_t)b0 * ZK;
  for (int half = 0; half < 2; ++half) {           // 8 tiles/wave in 2 groups
    floatx8 acc[4] = {};
    for (int kt = 0; kt < 16; ++kt) {
      bf16x16 a = load_a_g32(zrow + kt * 32, ZK, lane);
#pragma unroll
      for (int q = 0; q < 4; ++q)
        acc[q] = wmma_bf(a, load_b(A.P_z2h, (wv * 8 + half * 4 + q) * 16 + kt, lane),
                         acc[q]);
    }
#pragma unroll
    for (int q = 0; q < 4; ++q) {
      const int tIdx = wv * 8 + half * 4 + q;
      store_c_bf_bias(hT + tIdx * 16, TH, lane, acc[q], A.b_z2h + tIdx * 16);
    }
  }
  {
    floatx8 acc[2] = {};
    for (int kt = 0; kt < 16; ++kt) {
      bf16x16 a = load_a_g32(zrow + kt * 32, ZK, lane);
#pragma unroll
      for (int q = 0; q < 2; ++q)
        acc[q] = wmma_bf(a, load_b(A.P_z2i, (wv * 2 + q) * 16 + kt, lane), acc[q]);
    }
#pragma unroll
    for (int q = 0; q < 2; ++q)
      store_c_bf_bias(zin + (wv * 2 + q) * 16, ZI, lane, acc[q],
                      A.b_z2i + (wv * 2 + q) * 16);
  }
  __syncthreads();

  int curT = 0, curN = 0;
  for (int t = 0; t < NSTEP; ++t) {
    // ============ time_gru: x=[token|z_in](512), h=hT(1024) ============
    bf16* hOld = hT + curT * 16 * TH;
    bf16* hNew = hT + (curT ^ 1) * 16 * TH;
    for (int c = 0; c < 8; ++c) {                  // 128-col hidden chunks
      const int ntile = c * 8 + wv;                // 0..63
      {
        floatx8 acc[3] = {};
        for (int kt = 0; kt < 16; ++kt) {          // A loaded once, 3 gates
          const bf16* src = (kt < 8) ? (tok + kt * 32) : (zin + (kt - 8) * 32);
          bf16x16 a = load_a_lds(src, ZI, lane);
#pragma unroll
          for (int g = 0; g < 3; ++g)
            acc[g] = wmma_bf(a, load_b(A.P_time_ih, (g * 64 + ntile) * 16 + kt, lane),
                             acc[g]);
        }
#pragma unroll
        for (int g = 0; g < 3; ++g) store_c(gi + g * 2048 + wv * 16, 128, lane, acc[g]);
      }
      {
        floatx8 acc[3] = {};
        for (int kt = 0; kt < 32; ++kt) {
          bf16x16 a = load_a_lds(hOld + kt * 32, TH, lane);
#pragma unroll
          for (int g = 0; g < 3; ++g)
            acc[g] = wmma_bf(a, load_b(A.P_time_hh, (g * 64 + ntile) * 32 + kt, lane),
                             acc[g]);
        }
#pragma unroll
        for (int g = 0; g < 3; ++g) store_c(gh + g * 2048 + wv * 16, 128, lane, acc[g]);
      }
      __syncthreads();
      for (int e = tid; e < 2048; e += 256) {
        const int m = e >> 7, col = e & 127, gcol = c * 128 + col;
        const float ir = gi[m * 128 + col]         + sBtih[gcol];
        const float iz = gi[2048 + m * 128 + col]  + sBtih[TH + gcol];
        const float in_ = gi[4096 + m * 128 + col] + sBtih[2 * TH + gcol];
        const float hr = gh[m * 128 + col]         + sBthh[gcol];
        const float hz = gh[2048 + m * 128 + col]  + sBthh[TH + gcol];
        const float hn = gh[4096 + m * 128 + col]  + sBthh[2 * TH + gcol];
        const float r = sigm(ir + hr), zz = sigm(iz + hz);
        const float n = tanhf(in_ + r * hn);
        const float hp = (float)hOld[m * TH + gcol];
        hNew[m * TH + gcol] = (bf16)((1.0f - zz) * n + zz * hp);
      }
      __syncthreads();
    }
    curT ^= 1;
    bf16* hTime = hT + curT * 16 * TH;             // notes_summary

    // ============ notes_hid = time_to_notes(summary) ============
    {
      bf16* hDst = hN + curN * 16 * NH;
      floatx8 acc[4] = {};
      for (int kt = 0; kt < 32; ++kt) {
        bf16x16 a = load_a_lds(hTime + kt * 32, TH, lane);
#pragma unroll
        for (int q = 0; q < 4; ++q)
          acc[q] = wmma_bf(a, load_b(A.P_t2n, (wv * 4 + q) * 32 + kt, lane), acc[q]);
      }
#pragma unroll
      for (int q = 0; q < 4; ++q)
        store_c_bf_bias(hDst + (wv * 4 + q) * 16, NH, lane, acc[q],
                        sBt2n + (wv * 4 + q) * 16);
    }
    if (tid < 16) lens[tid] = 0;
    __syncthreads();

    // ============ inner decode: 15 steps ============
    for (int i = 0; i < NINNER; ++i) {
      bf16* hNo = hN + curN * 16 * NH;
      bf16* hNn = hN + (curN ^ 1) * 16 * NH;
      bf16* tokN = pred + i * (16 * NE);
      for (int c = 0; c < 4; ++c) {                // 128-col chunks of 512
        const int ntile = c * 8 + wv;              // 0..31
        {
          floatx8 acc[3] = {};
          for (int kt = 0; kt < 36; ++kt) {        // K=1024(summary)+128(token)
            const bf16* src = (kt < 32) ? (hTime + kt * 32) : (tokN + (kt - 32) * 32);
            const int ldk = (kt < 32) ? TH : NE;
            bf16x16 a = load_a_lds(src, ldk, lane);
#pragma unroll
            for (int g = 0; g < 3; ++g)
              acc[g] = wmma_bf(a, load_b(A.P_notes_ih, (g * 32 + ntile) * 36 + kt, lane),
                               acc[g]);
          }
#pragma unroll
          for (int g = 0; g < 3; ++g) store_c(gi + g * 2048 + wv * 16, 128, lane, acc[g]);
        }
        {
          floatx8 acc[3] = {};
          for (int kt = 0; kt < 16; ++kt) {
            bf16x16 a = load_a_lds(hNo + kt * 32, NH, lane);
#pragma unroll
            for (int g = 0; g < 3; ++g)
              acc[g] = wmma_bf(a, load_b(A.P_notes_hh, (g * 32 + ntile) * 16 + kt, lane),
                               acc[g]);
          }
#pragma unroll
          for (int g = 0; g < 3; ++g) store_c(gh + g * 2048 + wv * 16, 128, lane, acc[g]);
        }
        __syncthreads();
        for (int e = tid; e < 2048; e += 256) {
          const int m = e >> 7, col = e & 127, gcol = c * 128 + col;
          const float ir = gi[m * 128 + col]         + sBnih[gcol];
          const float iz = gi[2048 + m * 128 + col]  + sBnih[NH + gcol];
          const float in_ = gi[4096 + m * 128 + col] + sBnih[2 * NH + gcol];
          const float hr = gh[m * 128 + col]         + sBnhh[gcol];
          const float hz = gh[2048 + m * 128 + col]  + sBnhh[NH + gcol];
          const float hn = gh[4096 + m * 128 + col]  + sBnhh[2 * NH + gcol];
          const float r = sigm(ir + hr), zz = sigm(iz + hz);
          const float n = tanhf(in_ + r * hn);
          const float hp = (float)hNo[m * NH + gcol];
          hNn[m * NH + gcol] = (bf16)((1.0f - zz) * n + zz * hp);
        }
        __syncthreads();
      }
      curN ^= 1;
      bf16* hNow = hN + curN * 16 * NH;

      // ---- heads: pitch (9 tiles N=144) + dur (1 tile N=16), K=512 ----
      {
        floatx8 a0 = {}, a1 = {}, ad = {};
        for (int kt = 0; kt < 16; ++kt) {
          bf16x16 a = load_a_lds(hNow + kt * 32, NH, lane);
          a0 = wmma_bf(a, load_b(A.P_pitch, wv * 16 + kt, lane), a0);
          if (wv == 0) a1 = wmma_bf(a, load_b(A.P_pitch, 8 * 16 + kt, lane), a1);
          if (wv == 7) ad = wmma_bf(a, load_b(A.P_dur, kt, lane), ad);
        }
        store_c(head + wv * 16, PRP, lane, a0);
        if (wv == 0) store_c(head + 8 * 16, PRP, lane, a1);
        if (wv == 7) store_c(durs, 16, lane, ad);
      }
      __syncthreads();

      // ---- write logits to d_out (+bias) ----
      for (int e = tid; e < 16 * PR; e += 256) {
        const int m = e / PR, col = e % PR;
        const size_t o = ((((size_t)(b0 + m) * NSTEP + t) * NINNER + i) * PR) + col;
        A.out_pitch[o] = head[m * PRP + col] + sBpit[col];
      }
      for (int e = tid; e < 16 * 10; e += 256) {
        const int m = e / 10, col = e % 10;
        const size_t o = ((((size_t)(b0 + m) * NSTEP + t) * NINNER + i) * 10) + col;
        A.out_dur[o] = durs[m * 16 + col] + sBdur[col];
      }
      // ---- argmax + one-hot note token (one thread per row) ----
      if (tid < 16) {
        const int m = tid;
        float best = head[m * PRP] + sBpit[0];
        int bi_ = 0;
        for (int c2 = 1; c2 < PR; ++c2) {
          const float v = head[m * PRP + c2] + sBpit[c2];
          if (v > best) { best = v; bi_ = c2; }
        }
        if (bi_ == 129 && lens[m] == 0) lens[m] = i + 1;
        bf16* nt_ = ntok + m * NTK;
        for (int c2 = 0; c2 < NTK; ++c2) nt_[c2] = (bf16)0.0f;
        nt_[bi_] = (bf16)1.0f;
        for (int d = 0; d < 5; ++d) {
          const float v0 = durs[m * 16 + d * 2]     + sBdur[d * 2];
          const float v1 = durs[m * 16 + d * 2 + 1] + sBdur[d * 2 + 1];
          nt_[PR + d] = (bf16)((v1 > v0) ? 1.0f : 0.0f);
        }
      }
      __syncthreads();

      // ---- pred_emb = note_emb(token): N=128 (1 tile/wave), K=160 ----
      {
        floatx8 acc = {};
        for (int kt = 0; kt < 5; ++kt)
          acc = wmma_bf(load_a_lds(ntok + kt * 32, NTK, lane),
                        load_b(A.P_emb, wv * 5 + kt, lane), acc);
        store_c_bf_bias(pred + (i + 1) * (16 * NE) + wv * 16, NE, lane, acc,
                        sBemb + wv * 16);
      }
      __syncthreads();
    } // inner
    if (tid < 16 && lens[tid] == 0) lens[tid] = NINNER;
    __syncthreads();

    // ============ bidirectional emb GRU over 16 predicted embeddings ========
    for (int e = tid; e < 2 * 2 * 16 * EH; e += 256) hE[e] = (bf16)0.0f;
    __syncthreads();
    int curE = 0;
    for (int s = 0; s < 16; ++s) {
      for (int dir = 0; dir < 2; ++dir) {
        const int tt = dir ? (15 - s) : s;
        bf16* x  = pred + tt * (16 * NE);
        bf16* hO = hE + (dir * 2 + curE) * (16 * EH);
        bf16* hD = hE + (dir * 2 + (curE ^ 1)) * (16 * EH);
        const bf16* Pih = dir ? A.P_eb_ih : A.P_ef_ih;
        const bf16* Phh = dir ? A.P_eb_hh : A.P_ef_hh;
        const float* bih = sBe + dir * 768;
        const float* bhh = sBe + dir * 768 + 384;
        {
          floatx8 xa[3] = {}, ha[3] = {};
          for (int kt = 0; kt < 4; ++kt) {
            bf16x16 a = load_a_lds(x + kt * 32, NE, lane);
#pragma unroll
            for (int q = 0; q < 3; ++q)
              xa[q] = wmma_bf(a, load_b(Pih, (wv * 3 + q) * 4 + kt, lane), xa[q]);
          }
          for (int kt = 0; kt < 4; ++kt) {
            bf16x16 a = load_a_lds(hO + kt * 32, EH, lane);
#pragma unroll
            for (int q = 0; q < 3; ++q)
              ha[q] = wmma_bf(a, load_b(Phh, (wv * 3 + q) * 4 + kt, lane), ha[q]);
          }
#pragma unroll
          for (int q = 0; q < 3; ++q) {
            store_c(gi + (wv * 3 + q) * 16, 384, lane, xa[q]);
            store_c(gh + (wv * 3 + q) * 16, 384, lane, ha[q]);
          }
        }
        __syncthreads();
        for (int e = tid; e < 16 * EH; e += 256) {
          const int m = e >> 7, col = e & 127;
          const float ir = gi[m * 384 + col]       + bih[col];
          const float iz = gi[m * 384 + 128 + col] + bih[128 + col];
          const float in_ = gi[m * 384 + 256 + col] + bih[256 + col];
          const float hr = gh[m * 384 + col]       + bhh[col];
          const float hz = gh[m * 384 + 128 + col] + bhh[128 + col];
          const float hn = gh[m * 384 + 256 + col] + bhh[256 + col];
          const float r = sigm(ir + hr), zz = sigm(iz + hz);
          const float n = tanhf(in_ + r * hn);
          const float hp = (float)hO[m * EH + col];
          const float hv = (1.0f - zz) * n + zz * hp;
          hD[m * EH + col] = (tt < lens[m]) ? (bf16)hv : (bf16)hp;
        }
        __syncthreads();
      }
      curE ^= 1;
    }
    // token = [h_fw_final | h_bw_final]
    for (int e = tid; e < 16 * EH; e += 256) {
      const int m = e >> 7, col = e & 127;
      tok[m * ZI + col]       = hE[(0 * 2 + curE) * (16 * EH) + m * EH + col];
      tok[m * ZI + 128 + col] = hE[(1 * 2 + curE) * (16 * EH) + m * EH + col];
    }
    __syncthreads();
  } // outer
}

// ---------------- host side ----------------
extern "C" void kernel_launch(void* const* d_in, const int* in_sizes, int n_in,
                              void* d_out, int out_size, void* d_ws, size_t ws_size,
                              hipStream_t stream) {
  (void)in_sizes; (void)n_in; (void)out_size;
  bf16* ws = (bf16*)d_ws;
  size_t off = 0;
  auto take = [&](size_t n) { size_t r = off; off += (n + 255) & ~(size_t)255; return r; };
  const size_t oTih = take((size_t)3072 * 512);
  const size_t oThh = take((size_t)3072 * 1024);
  const size_t oZ2h = take((size_t)1024 * 512);
  const size_t oZ2i = take((size_t)256 * 512);
  const size_t oT2n = take((size_t)512 * 1024);
  const size_t oNih = take((size_t)1536 * 1152);
  const size_t oNhh = take((size_t)1536 * 512);
  const size_t oPit = take((size_t)144 * 512);
  const size_t oDur = take((size_t)16 * 512);
  const size_t oEmb = take((size_t)128 * 160);
  const size_t oFih = take((size_t)384 * 128);
  const size_t oFhh = take((size_t)384 * 128);
  const size_t oBih = take((size_t)384 * 128);
  const size_t oBhh = take((size_t)384 * 128);
  if (ws_size < off * sizeof(bf16)) return;

  auto pack = [&](int in_idx, size_t dst, int O, int I, int Np, int Kp) {
    const int total = Np * Kp;
    hipLaunchKernelGGL(pack_w, dim3((total + 255) / 256), dim3(256), 0, stream,
                       (const float*)d_in[in_idx], ws + dst, O, I, Np, Kp);
  };
  pack(13, oTih, 3072, 512, 3072, 512);    // time_gru.w_ih
  pack(14, oThh, 3072, 1024, 3072, 1024);  // time_gru.w_hh
  pack(3,  oZ2h, 1024, 512, 1024, 512);    // z2dec_hid.w
  pack(5,  oZ2i, 256, 512, 256, 512);      // z2dec_in.w
  pack(7,  oT2n, 512, 1024, 512, 1024);    // time_to_notes.w
  pack(17, oNih, 1536, 1152, 1536, 1152);  // notes_gru.w_ih
  pack(18, oNhh, 1536, 512, 1536, 512);    // notes_gru.w_hh
  pack(9,  oPit, 130, 512, 144, 512);      // pitch_out.w
  pack(11, oDur, 10, 512, 16, 512);        // dur_out.w
  pack(1,  oEmb, 128, 135, 128, 160);      // note_emb.w
  pack(21, oFih, 384, 128, 384, 128);      // emb_gru_fw.w_ih
  pack(22, oFhh, 384, 128, 384, 128);      // emb_gru_fw.w_hh
  pack(25, oBih, 384, 128, 384, 128);      // emb_gru_bw.w_ih
  pack(26, oBhh, 384, 128, 384, 128);      // emb_gru_bw.w_hh

  Args A;
  A.z        = (const float*)d_in[0];
  A.nw       = (const float*)d_in[1];
  A.nb       = (const float*)d_in[2];
  A.b_z2h    = (const float*)d_in[4];
  A.b_z2i    = (const float*)d_in[6];
  A.b_t2n    = (const float*)d_in[8];
  A.b_pitch  = (const float*)d_in[10];
  A.b_dur    = (const float*)d_in[12];
  A.t_bih    = (const float*)d_in[15];
  A.t_bhh    = (const float*)d_in[16];
  A.n_bih    = (const float*)d_in[19];
  A.n_bhh    = (const float*)d_in[20];
  A.ef_bih   = (const float*)d_in[23];
  A.ef_bhh   = (const float*)d_in[24];
  A.eb_bih   = (const float*)d_in[27];
  A.eb_bhh   = (const float*)d_in[28];
  A.dec_init = (const float*)d_in[29];
  A.b_emb    = A.nb;
  A.P_time_ih = ws + oTih;  A.P_time_hh = ws + oThh;
  A.P_z2h = ws + oZ2h;      A.P_z2i = ws + oZ2i;      A.P_t2n = ws + oT2n;
  A.P_notes_ih = ws + oNih; A.P_notes_hh = ws + oNhh;
  A.P_pitch = ws + oPit;    A.P_dur = ws + oDur;      A.P_emb = ws + oEmb;
  A.P_ef_ih = ws + oFih;    A.P_ef_hh = ws + oFhh;
  A.P_eb_ih = ws + oBih;    A.P_eb_hh = ws + oBhh;
  A.out_pitch = (float*)d_out;
  A.out_dur   = (float*)d_out + (size_t)NB * NSTEP * NINNER * PR;

  hipFuncSetAttribute((const void*)ptd_main,
                      hipFuncAttributeMaxDynamicSharedMemorySize, (int)SMEM_BYTES);
  hipLaunchKernelGGL(ptd_main, dim3(NB / 16), dim3(256), SMEM_BYTES, stream, A);
}
